// GCTMultiLabel_81939386073702
// MI455X (gfx1250) — matrix operand order repaired and checked by reading, not compiled, not run on hardware
//
#include <hip/hip_runtime.h>

#define N_NODES 2048
#define IN_DIM_ 64
#define HID     1024
#define HEADS   16
#define D_HEAD_ 64
#define LAYERS_ 2
#define NE      65536

typedef __attribute__((ext_vector_type(16))) __bf16       bf16x16;
typedef __attribute__((ext_vector_type(8)))  float        f32x8;
typedef __attribute__((ext_vector_type(4)))  unsigned int u32x4;

union Frag { bf16x16 v; u32x4 u[2]; };

#if __has_builtin(__builtin_amdgcn_global_load_async_to_lds_b128) && \
    __has_builtin(__builtin_amdgcn_s_wait_asynccnt)
#define USE_ASYNC_LDS 1
#else
#define USE_ASYNC_LDS 0
#endif

#if USE_ASYNC_LDS
typedef int i32x4g __attribute__((vector_size(16)));   // matches builtin pointee type
__device__ __forceinline__ void async_g2l_b128(const void* g, void* l) {
  __builtin_amdgcn_global_load_async_to_lds_b128(
      (__attribute__((address_space(1))) i32x4g*)g,
      (__attribute__((address_space(3))) i32x4g*)l, 0, 0);
}
__device__ __forceinline__ void async_wait0() {
  __builtin_amdgcn_s_wait_asynccnt(0);
}
#else
__device__ __forceinline__ void async_wait0() {}
#endif

__device__ __forceinline__ unsigned short f2bf(float f) {
  union { float f; unsigned int u; } x; x.f = f;
  unsigned int u = x.u;
  u += 0x7fffu + ((u >> 16) & 1u);   // round-to-nearest-even
  return (unsigned short)(u >> 16);
}

// ---------------------------------------------------------------- utilities
__global__ void zero_kernel(float* __restrict__ p, int n) {
  int i = blockIdx.x * blockDim.x + threadIdx.x;
  int stride = gridDim.x * blockDim.x;
  for (; i < n; i += stride) p[i] = 0.0f;
}

__global__ void cvt_kernel(const float* __restrict__ s, unsigned short* __restrict__ d, int n) {
  int i = blockIdx.x * blockDim.x + threadIdx.x;
  int stride = gridDim.x * blockDim.x;
  for (; i < n; i += stride) d[i] = f2bf(s[i]);
}

// deterministic scatter: pass 1 picks a winner edge per slot, pass 2 writes it
__global__ void scatter1_kernel(const int* __restrict__ ei, unsigned int* __restrict__ winner) {
  int e = blockIdx.x * blockDim.x + threadIdx.x;
  if (e < NE) {
    int s = ei[e], d = ei[NE + e];
    atomicMax(&winner[(size_t)s * N_NODES + d], (unsigned int)(e + 1));
  }
}
__global__ void scatter2_kernel(const int* __restrict__ ei, const float* __restrict__ ew,
                                const float* __restrict__ ec, const unsigned int* __restrict__ winner,
                                float2* __restrict__ WC) {
  int e = blockIdx.x * blockDim.x + threadIdx.x;
  if (e < NE) {
    int s = ei[e], d = ei[NE + e];
    size_t idx = (size_t)s * N_NODES + d;
    if (winner[idx] == (unsigned int)(e + 1)) WC[idx] = make_float2(ew[e], ec[e]);
  }
}

// ---------------------------------------------------------------- WMMA GEMM
// C = A[MxK](bf16) * B[KxN](bf16) with fused epilogues. Block 256 thr = 8 waves,
// tile 128x128x32, wave tile 32x64 (2x4 WMMA accumulators).
#define GM_BIAS_F32       0
#define GM_BIAS_F32_BF16  1
#define GM_BF16           2
#define GM_BF16_T         3
#define GM_BIAS_RELU_BF16 4

#define BM 128
#define BN 128
#define BK 32

template <int MODE>
__global__ __launch_bounds__(256) void gemm_bf16_kernel(
    const unsigned short* __restrict__ A, const unsigned short* __restrict__ B,
    const float* __restrict__ bias, float* __restrict__ Cf,
    unsigned short* __restrict__ Cbf, int M, int Nn, int K) {
  __shared__ __align__(16) unsigned short As[2][BM][BK + 8];   // pad: 40 halfs/row
  __shared__ __align__(16) unsigned short Bs[2][BK][BN + 8];   // pad: 136 halfs/row

  const int tid = threadIdx.x;
  const int m0 = blockIdx.y * BM;
  const int n0 = blockIdx.x * BN;
  const int wv = tid >> 5;
  const int wm = wv >> 1;      // 0..3  (M direction)
  const int wn = wv & 1;       // 0..1  (N direction)
  const int lane = tid & 31;
  const int lh = lane >> 4;
  const int lm = lane & 15;

  f32x8 zero8 = {0.f, 0.f, 0.f, 0.f, 0.f, 0.f, 0.f, 0.f};
  f32x8 acc[2][4];
#pragma unroll
  for (int i = 0; i < 2; ++i)
#pragma unroll
    for (int j = 0; j < 4; ++j) acc[i][j] = zero8;

  auto load_tile = [&](int buf, int ks) {
#pragma unroll
    for (int it = 0; it < 2; ++it) {        // A: 128 rows x 32 halfs = 512 x 16B
      int c = tid + it * 256;
      int row = c >> 2, cc = c & 3;
      const unsigned short* src = A + (size_t)(m0 + row) * K + ks + cc * 8;
      unsigned short* dst = &As[buf][row][cc * 8];
#if USE_ASYNC_LDS
      async_g2l_b128(src, dst);
#else
      *(u32x4*)dst = *(const u32x4*)src;
#endif
    }
#pragma unroll
    for (int it = 0; it < 2; ++it) {        // B: 32 rows x 128 halfs = 512 x 16B
      int c = tid + it * 256;
      int row = c >> 4, cc = c & 15;
      const unsigned short* src = B + (size_t)(ks + row) * Nn + n0 + cc * 8;
      unsigned short* dst = &Bs[buf][row][cc * 8];
#if USE_ASYNC_LDS
      async_g2l_b128(src, dst);
#else
      *(u32x4*)dst = *(const u32x4*)src;
#endif
    }
  };

  load_tile(0, 0);
  async_wait0();
  __syncthreads();
  const int nk = K / BK;
  for (int kb = 0; kb < nk; ++kb) {
    int buf = kb & 1;
    if (kb + 1 < nk) load_tile(buf ^ 1, (kb + 1) * BK);
    if (kb + 2 < nk) {  // gfx1250 global_prefetch_b8 path
      __builtin_prefetch(A + (size_t)(m0 + (tid >> 1)) * K + (kb + 2) * BK, 0, 1);
      __builtin_prefetch(B + (size_t)((kb + 2) * BK + (tid >> 3)) * Nn + n0, 0, 1);
    }

    Frag af[2];
#pragma unroll
    for (int mt = 0; mt < 2; ++mt) {
      const unsigned short* ap = &As[buf][wm * 32 + mt * 16 + lm][lh * 8];
      af[mt].u[0] = *(const u32x4*)ap;          // k = lh*8 .. +7
      af[mt].u[1] = *(const u32x4*)(ap + 16);   // k = lh*8+16 .. +23
    }
#pragma unroll
    for (int nt = 0; nt < 4; ++nt) {
      Frag bfr;
      const int bcol = wn * 64 + nt * 16 + lm;
#pragma unroll
      for (int j = 0; j < 8; ++j) {
        int k0 = lh * 8 + ((2 * j) & 7) + ((2 * j) >> 3) * 16;
        bfr.u[j >> 2][j & 3] = (unsigned)Bs[buf][k0][bcol] | ((unsigned)Bs[buf][k0 + 1][bcol] << 16);
      }
#pragma unroll
      for (int mt = 0; mt < 2; ++mt) {
        acc[mt][nt] = __builtin_amdgcn_wmma_f32_16x16x32_bf16(
            false, af[mt].v, false, bfr.v, (short)0, acc[mt][nt], false, false);
      }
    }
    async_wait0();
    __syncthreads();
  }

#pragma unroll
  for (int nt = 0; nt < 4; ++nt) {
    int col = n0 + wn * 64 + nt * 16 + lm;
    float bv = 0.0f;
    if (MODE == GM_BIAS_F32 || MODE == GM_BIAS_F32_BF16 || MODE == GM_BIAS_RELU_BF16)
      bv = bias[col];
#pragma unroll
    for (int mt = 0; mt < 2; ++mt) {
#pragma unroll
      for (int r = 0; r < 8; ++r) {
        int row = m0 + wm * 32 + mt * 16 + r + lh * 8;
        float v = acc[mt][nt][r] + bv;
        if (MODE == GM_BIAS_RELU_BF16) v = fmaxf(v, 0.0f);
        if (MODE == GM_BIAS_F32 || MODE == GM_BIAS_F32_BF16)
          Cf[(size_t)row * Nn + col] = v;
        if (MODE == GM_BF16 || MODE == GM_BIAS_F32_BF16 || MODE == GM_BIAS_RELU_BF16)
          Cbf[(size_t)row * Nn + col] = f2bf(v);
        if (MODE == GM_BF16_T)
          Cbf[(size_t)col * M + row] = f2bf(v);
      }
    }
  }
}

// ------------------------------------------------------- fused flash attention
// grid (N/64, HEADS), 128 thr = 4 waves; each wave owns one (head, 16-row) tile
// and streams keys 64 at a time (4 m-subtiles) to amortize softmax reductions.
__global__ __launch_bounds__(128) void attn_kernel(
    const unsigned short* __restrict__ Q,   // [N, HID] bf16
    const unsigned short* __restrict__ Kt,  // [HID, N] bf16 (pre-transposed)
    const unsigned short* __restrict__ V,   // [N, HID] bf16
    const float2* __restrict__ WC,          // [N, N] interleaved (W, C)
    const float* __restrict__ bw, const float* __restrict__ bc,
    unsigned short* __restrict__ AO)        // [N, HID] bf16
{
  __shared__ __align__(16) unsigned short Ps[4][16][72];  // per-wave 16x64 P tile

  const int tid = threadIdx.x;
  const int w = tid >> 5;
  const int lane = tid & 31;
  const int lh = lane >> 4, lm = lane & 15;
  const int head = blockIdx.y;
  const int n0 = (blockIdx.x * 4 + w) * 16;

  const float bwh = bw[head], bch = bc[head];
  const float scale = 0.125f;  // 1/sqrt(64)

  Frag qa[2];
#pragma unroll
  for (int kc = 0; kc < 2; ++kc) {
    const unsigned short* qp = Q + (size_t)(n0 + lm) * HID + head * 64 + kc * 32 + lh * 8;
    qa[kc].u[0] = *(const u32x4*)qp;
    qa[kc].u[1] = *(const u32x4*)(qp + 16);
  }

  f32x8 zero8 = {0.f, 0.f, 0.f, 0.f, 0.f, 0.f, 0.f, 0.f};
  f32x8 o[4];
#pragma unroll
  for (int vt = 0; vt < 4; ++vt) o[vt] = zero8;
  float rmax[8], rsum[8];
#pragma unroll
  for (int r = 0; r < 8; ++r) { rmax[r] = -3.0e38f; rsum[r] = 0.0f; }

  for (int m0 = 0; m0 < N_NODES; m0 += 64) {
    // S = Q Kt  (four 16x16 m-subtiles, K=64 split into two K=32 WMMAs)
    f32x8 s[4];
#pragma unroll
    for (int mt = 0; mt < 4; ++mt) s[mt] = zero8;
#pragma unroll
    for (int mt = 0; mt < 4; ++mt) {
#pragma unroll
      for (int kc = 0; kc < 2; ++kc) {
        Frag kb;
        const unsigned short* kp =
            Kt + (size_t)(head * 64 + kc * 32) * N_NODES + m0 + mt * 16 + lm;
#pragma unroll
        for (int j = 0; j < 8; ++j) {
          int k0 = lh * 8 + ((2 * j) & 7) + ((2 * j) >> 3) * 16;
          kb.u[j >> 2][j & 3] = (unsigned)kp[(size_t)k0 * N_NODES] | ((unsigned)kp[(size_t)(k0 + 1) * N_NODES] << 16);
        }
        s[mt] = __builtin_amdgcn_wmma_f32_16x16x32_bf16(
            false, qa[kc].v, false, kb.v, (short)0, s[mt], false, false);
      }
    }
    // logits = s*scale + bw*W + bc*C  (one b64 load per element from packed WC)
    float l[4][8];
#pragma unroll
    for (int mt = 0; mt < 4; ++mt)
#pragma unroll
      for (int r = 0; r < 8; ++r) {
        int row = n0 + r + lh * 8;
        int col = m0 + mt * 16 + lm;
        float2 wc = WC[(size_t)row * N_NODES + col];
        l[mt][r] = s[mt][r] * scale + bwh * wc.x + bch * wc.y;
      }
    // online softmax: row reductions over the 16 lanes of each half-wave
    float t[8];
#pragma unroll
    for (int r = 0; r < 8; ++r)
      t[r] = fmaxf(fmaxf(l[0][r], l[1][r]), fmaxf(l[2][r], l[3][r]));
#pragma unroll
    for (int offd = 1; offd <= 8; offd <<= 1)
#pragma unroll
      for (int r = 0; r < 8; ++r) t[r] = fmaxf(t[r], __shfl_xor(t[r], offd, 32));
    float al[8];
#pragma unroll
    for (int r = 0; r < 8; ++r) {
      float nm = fmaxf(rmax[r], t[r]);
      al[r] = __expf(rmax[r] - nm);
      rmax[r] = nm;
    }
#pragma unroll
    for (int mt = 0; mt < 4; ++mt)
#pragma unroll
      for (int r = 0; r < 8; ++r) l[mt][r] = __expf(l[mt][r] - rmax[r]);
    float u[8];
#pragma unroll
    for (int r = 0; r < 8; ++r)
      u[r] = (l[0][r] + l[1][r]) + (l[2][r] + l[3][r]);
#pragma unroll
    for (int offd = 1; offd <= 8; offd <<= 1)
#pragma unroll
      for (int r = 0; r < 8; ++r) u[r] += __shfl_xor(u[r], offd, 32);
#pragma unroll
    for (int r = 0; r < 8; ++r) rsum[r] = rsum[r] * al[r] + u[r];
#pragma unroll
    for (int vt = 0; vt < 4; ++vt)
#pragma unroll
      for (int r = 0; r < 8; ++r) o[vt][r] *= al[r];
    // route P (C-layout) through LDS to A-fragment layout
#pragma unroll
    for (int mt = 0; mt < 4; ++mt)
#pragma unroll
      for (int r = 0; r < 8; ++r)
        Ps[w][r + lh * 8][mt * 16 + lm] = f2bf(l[mt][r]);
    Frag pa[2];
#pragma unroll
    for (int f = 0; f < 2; ++f) {
      const unsigned short* pp = &Ps[w][lm][f * 32 + lh * 8];
      pa[f].u[0] = *(const u32x4*)pp;
      pa[f].u[1] = *(const u32x4*)(pp + 16);
    }
    // O += P V
#pragma unroll
    for (int f = 0; f < 2; ++f) {
#pragma unroll
      for (int vt = 0; vt < 4; ++vt) {
        Frag vb;
        const unsigned short* vp =
            V + (size_t)(m0 + f * 32) * HID + head * 64 + vt * 16 + lm;
#pragma unroll
        for (int j = 0; j < 8; ++j) {
          int k0 = lh * 8 + ((2 * j) & 7) + ((2 * j) >> 3) * 16;
          vb.u[j >> 2][j & 3] = (unsigned)vp[(size_t)k0 * HID] | ((unsigned)vp[(size_t)(k0 + 1) * HID] << 16);
        }
        o[vt] = __builtin_amdgcn_wmma_f32_16x16x32_bf16(
            false, pa[f].v, false, vb.v, (short)0, o[vt], false, false);
      }
    }
  }
#pragma unroll
  for (int vt = 0; vt < 4; ++vt)
#pragma unroll
    for (int r = 0; r < 8; ++r) {
      int row = n0 + r + lh * 8;
      int col = head * 64 + vt * 16 + lm;
      AO[(size_t)row * HID + col] = f2bf(o[vt][r] / rsum[r]);
    }
}

// ---------------------------------------------------------- LayerNorm(h + d)
__global__ __launch_bounds__(256) void ln_kernel(
    const float* __restrict__ hin, const float* __restrict__ dl,
    const float* __restrict__ sc, const float* __restrict__ bi,
    float* __restrict__ hout, unsigned short* __restrict__ hbf) {
  __shared__ float sred[16];
  const int row = blockIdx.x;
  const int tid = threadIdx.x;
  float v[4], s = 0.f, s2 = 0.f;
#pragma unroll
  for (int i = 0; i < 4; ++i) {
    int c = tid + i * 256;
    float xv = hin[(size_t)row * HID + c] + dl[(size_t)row * HID + c];
    v[i] = xv; s += xv; s2 += xv * xv;
  }
#pragma unroll
  for (int offd = 16; offd > 0; offd >>= 1) {
    s += __shfl_xor(s, offd, 32);
    s2 += __shfl_xor(s2, offd, 32);
  }
  if ((tid & 31) == 0) { sred[tid >> 5] = s; sred[8 + (tid >> 5)] = s2; }
  __syncthreads();
  float S = 0.f, S2 = 0.f;
#pragma unroll
  for (int i = 0; i < 8; ++i) { S += sred[i]; S2 += sred[8 + i]; }
  float mu = S * (1.0f / HID);
  float var = S2 * (1.0f / HID) - mu * mu;
  float rstd = rsqrtf(var + 1e-5f);
#pragma unroll
  for (int i = 0; i < 4; ++i) {
    int c = tid + i * 256;
    float y = (v[i] - mu) * rstd * sc[c] + bi[c];
    hout[(size_t)row * HID + c] = y;
    hbf[(size_t)row * HID + c] = f2bf(y);
  }
}

// --------------------------------------------------------------- final head
__global__ __launch_bounds__(256) void out_kernel(
    const float* __restrict__ h, const float* __restrict__ ow,
    const float* __restrict__ ob, float* __restrict__ outp) {
  __shared__ float sred[8];
  const int row = blockIdx.x;
  const int tid = threadIdx.x;
  float s = 0.f;
#pragma unroll
  for (int i = 0; i < 4; ++i) {
    int c = tid + i * 256;
    s += h[(size_t)row * HID + c] * ow[c];
  }
#pragma unroll
  for (int offd = 16; offd > 0; offd >>= 1) s += __shfl_xor(s, offd, 32);
  if ((tid & 31) == 0) sred[tid >> 5] = s;
  __syncthreads();
  if (tid == 0) {
    float S = 0.f;
#pragma unroll
    for (int i = 0; i < 8; ++i) S += sred[i];
    outp[row] = S + ob[0];
  }
}

// ---------------------------------------------------------------- launcher
extern "C" void kernel_launch(void* const* d_in, const int* in_sizes, int n_in,
                              void* d_out, int out_size, void* d_ws, size_t ws_size,
                              hipStream_t stream) {
  (void)in_sizes; (void)n_in; (void)out_size; (void)ws_size;
  const float* x      = (const float*)d_in[0];
  const int*   ei     = (const int*)  d_in[1];
  const float* ew     = (const float*)d_in[2];
  const float* ec     = (const float*)d_in[3];
  const float* node_w = (const float*)d_in[4];
  const float* node_b = (const float*)d_in[5];
  const float* out_w  = (const float*)d_in[6];
  const float* out_b  = (const float*)d_in[7];
  const float* qw  = (const float*)d_in[8];
  const float* kw  = (const float*)d_in[9];
  const float* vw  = (const float*)d_in[10];
  const float* ow  = (const float*)d_in[11];
  const float* ob  = (const float*)d_in[12];
  const float* bw  = (const float*)d_in[13];
  const float* bc  = (const float*)d_in[14];
  const float* n1s = (const float*)d_in[15];
  const float* n1b = (const float*)d_in[16];
  const float* n2s = (const float*)d_in[17];
  const float* n2b = (const float*)d_in[18];
  const float* f1w = (const float*)d_in[19];
  const float* f1b = (const float*)d_in[20];
  const float* f2w = (const float*)d_in[21];
  const float* f2b = (const float*)d_in[22];
  float* outp = (float*)d_out;

  char* ws = (char*)d_ws;
  size_t off = 0;
  auto alloc = [&](size_t bytes) -> void* {
    void* p = ws + off;
    off = (off + bytes + 255) & ~(size_t)255;
    return p;
  };

  const size_t NN = (size_t)N_NODES * N_NODES;
  float2* WC = (float2*)alloc(3 * NN * sizeof(float));  // interleaved W,C | winner
  unsigned int* winner = (unsigned int*)((float*)WC + 2 * NN);
  float* h   = (float*)alloc((size_t)N_NODES * HID * sizeof(float));
  float* tmp = (float*)alloc((size_t)N_NODES * HID * sizeof(float));
  unsigned short* hbf  = (unsigned short*)alloc((size_t)N_NODES * HID * 2);
  unsigned short* qbf  = (unsigned short*)alloc((size_t)N_NODES * HID * 2);
  unsigned short* ktbf = (unsigned short*)alloc((size_t)HID * N_NODES * 2);
  unsigned short* vbf  = (unsigned short*)alloc((size_t)N_NODES * HID * 2);
  unsigned short* aobf = (unsigned short*)alloc((size_t)N_NODES * HID * 2);
  unsigned short* fhbf = (unsigned short*)alloc((size_t)N_NODES * 2 * HID * 2);
  unsigned short* xbf  = (unsigned short*)alloc((size_t)N_NODES * IN_DIM_ * 2);
  unsigned short* nwbf = (unsigned short*)alloc((size_t)IN_DIM_ * HID * 2);
  unsigned short* wqb  = (unsigned short*)alloc((size_t)HID * HID * 2);
  unsigned short* wkb  = (unsigned short*)alloc((size_t)HID * HID * 2);
  unsigned short* wvb  = (unsigned short*)alloc((size_t)HID * HID * 2);
  unsigned short* wob  = (unsigned short*)alloc((size_t)HID * HID * 2);
  unsigned short* wf1b = (unsigned short*)alloc((size_t)HID * 2 * HID * 2);
  unsigned short* wf2b = (unsigned short*)alloc((size_t)2 * HID * HID * 2);

  // Build packed edge-bias matrix (deterministic winner via atomicMax)
  zero_kernel<<<dim3(4096), dim3(256), 0, stream>>>((float*)WC, (int)(3 * NN));
  scatter1_kernel<<<dim3(NE / 256), dim3(256), 0, stream>>>(ei, winner);
  scatter2_kernel<<<dim3(NE / 256), dim3(256), 0, stream>>>(ei, ew, ec, winner, WC);

  // h = x @ node_w + node_b
  cvt_kernel<<<dim3(512), dim3(256), 0, stream>>>(x, xbf, N_NODES * IN_DIM_);
  cvt_kernel<<<dim3(512), dim3(256), 0, stream>>>(node_w, nwbf, IN_DIM_ * HID);
  gemm_bf16_kernel<GM_BIAS_F32_BF16>
      <<<dim3(HID / BN, N_NODES / BM), dim3(256), 0, stream>>>(
          xbf, nwbf, node_b, h, hbf, N_NODES, HID, IN_DIM_);

  for (int l = 0; l < LAYERS_; ++l) {
    cvt_kernel<<<dim3(1024), dim3(256), 0, stream>>>(qw + (size_t)l * HID * HID, wqb, HID * HID);
    cvt_kernel<<<dim3(1024), dim3(256), 0, stream>>>(kw + (size_t)l * HID * HID, wkb, HID * HID);
    cvt_kernel<<<dim3(1024), dim3(256), 0, stream>>>(vw + (size_t)l * HID * HID, wvb, HID * HID);
    cvt_kernel<<<dim3(1024), dim3(256), 0, stream>>>(ow + (size_t)l * HID * HID, wob, HID * HID);
    cvt_kernel<<<dim3(2048), dim3(256), 0, stream>>>(f1w + (size_t)l * HID * 2 * HID, wf1b, HID * 2 * HID);
    cvt_kernel<<<dim3(2048), dim3(256), 0, stream>>>(f2w + (size_t)l * 2 * HID * HID, wf2b, 2 * HID * HID);

    gemm_bf16_kernel<GM_BF16><<<dim3(HID / BN, N_NODES / BM), dim3(256), 0, stream>>>(
        hbf, wqb, nullptr, nullptr, qbf, N_NODES, HID, HID);
    gemm_bf16_kernel<GM_BF16_T><<<dim3(HID / BN, N_NODES / BM), dim3(256), 0, stream>>>(
        hbf, wkb, nullptr, nullptr, ktbf, N_NODES, HID, HID);
    gemm_bf16_kernel<GM_BF16><<<dim3(HID / BN, N_NODES / BM), dim3(256), 0, stream>>>(
        hbf, wvb, nullptr, nullptr, vbf, N_NODES, HID, HID);

    attn_kernel<<<dim3(N_NODES / 64, HEADS), dim3(128), 0, stream>>>(
        qbf, ktbf, vbf, WC, bw + (size_t)l * HEADS, bc + (size_t)l * HEADS, aobf);

    gemm_bf16_kernel<GM_BIAS_F32><<<dim3(HID / BN, N_NODES / BM), dim3(256), 0, stream>>>(
        aobf, wob, ob + (size_t)l * HID, tmp, nullptr, N_NODES, HID, HID);
    ln_kernel<<<dim3(N_NODES), dim3(256), 0, stream>>>(
        h, tmp, n1s + (size_t)l * HID, n1b + (size_t)l * HID, h, hbf);

    gemm_bf16_kernel<GM_BIAS_RELU_BF16>
        <<<dim3(2 * HID / BN, N_NODES / BM), dim3(256), 0, stream>>>(
            hbf, wf1b, f1b + (size_t)l * 2 * HID, nullptr, fhbf, N_NODES, 2 * HID, HID);
    gemm_bf16_kernel<GM_BIAS_F32><<<dim3(HID / BN, N_NODES / BM), dim3(256), 0, stream>>>(
        fhbf, wf2b, f2b + (size_t)l * HID, tmp, nullptr, N_NODES, HID, 2 * HID);
    ln_kernel<<<dim3(N_NODES), dim3(256), 0, stream>>>(
        h, tmp, n2s + (size_t)l * HID, n2b + (size_t)l * HID, h, hbf);
  }

  out_kernel<<<dim3(N_NODES), dim3(256), 0, stream>>>(h, out_w, out_b, outp);
}
